// TransformerBlock_42838003810661
// MI455X (gfx1250) — compile-verified, matrix operand order checked by reading, not run on hardware
//
#include <hip/hip_runtime.h>

#define DIMX   1024
#define HEADS  16
#define DH     64
#define SEQ    2048
#define BATCH  4
#define ROWS   (BATCH * SEQ)     // 8192
#define BHN    (BATCH * HEADS)   // 64

typedef unsigned short u16;
typedef __attribute__((ext_vector_type(16))) __bf16 v16bf;
typedef __attribute__((ext_vector_type(8)))  float  v8f;

#ifndef __has_builtin
#define __has_builtin(x) 0
#endif

#if __has_builtin(__builtin_amdgcn_global_load_async_to_lds_b128) && \
    __has_builtin(__builtin_amdgcn_s_wait_asynccnt)
#define USE_ASYNC 1
#else
#define USE_ASYNC 0
#endif

#if USE_ASYNC
typedef int v4i_t __attribute__((vector_size(16)));
typedef __attribute__((address_space(1))) v4i_t gv4i;
typedef __attribute__((address_space(3))) v4i_t lv4i;
#define WAIT_ASYNC(n) __builtin_amdgcn_s_wait_asynccnt(n)
#else
#define WAIT_ASYNC(n)
#endif

// copy 16 bytes global -> LDS (async on CDNA5, tracked by ASYNCcnt)
__device__ __forceinline__ void cp16(u16* l, const u16* g) {
#if USE_ASYNC
    __builtin_amdgcn_global_load_async_to_lds_b128(
        (gv4i*)(unsigned long long)g,
        (lv4i*)(unsigned)(unsigned long long)l,
        0, 0);
#else
    *(uint4*)l = *(const uint4*)g;
#endif
}

#define WMMA_BF16(a, b, c) \
    __builtin_amdgcn_wmma_f32_16x16x32_bf16(false, (a), false, (b), (short)0, (c), false, false)

union FragU {
    v16bf v;
    uint4 u[2];
};

// float -> bf16 round-to-nearest-even
__device__ __forceinline__ u16 f2bf(float x) {
    union { float f; unsigned u; } c; c.f = x;
    unsigned r = c.u + 0x7FFFu + ((c.u >> 16) & 1u);
    return (u16)(r >> 16);
}

// 16x32 bf16 fragment, CDNA5 WMMA A/B layout.
// lane l (l<16): row l, k = 0..7 then 16..23 ; lane l+16: row l, k = 8..15 then 24..31.
__device__ __forceinline__ v16bf load_frag(const u16* base, int stride, int lane) {
    const int r    = lane & 15;
    const int koff = (lane >> 4) * 8;
    const u16* p = base + (size_t)r * stride + koff;
    FragU f;
    f.u[0] = *(const uint4*)(p);
    f.u[1] = *(const uint4*)(p + 16);
    return f.v;
}

// ---------------- conversion kernels ----------------

__global__ void cvt_bf16_vec4(const float* __restrict__ in, u16* __restrict__ out, int n4) {
    int i = blockIdx.x * blockDim.x + threadIdx.x;
    if (i < n4) {
        float4 v = ((const float4*)in)[i];
        uint2 p;
        p.x = (unsigned)f2bf(v.x) | ((unsigned)f2bf(v.y) << 16);
        p.y = (unsigned)f2bf(v.z) | ((unsigned)f2bf(v.w) << 16);
        ((uint2*)out)[i] = p;
    }
}

// in: [R][C] f32 row-major -> out: [C][R] bf16 (transposed)
__global__ void transpose_cvt(const float* __restrict__ in, u16* __restrict__ out, int R, int C) {
    int i = blockIdx.x * blockDim.x + threadIdx.x;
    if (i < R * C) {
        int n  = i / R;
        int kk = i - n * R;
        out[i] = f2bf(in[(size_t)kk * C + n]);
    }
}

// ---------------- GEMM core: 256 threads / 8 waves -> 128(M) x 64(N) block tile ----------------
// A: [M][K] bf16 row-major (ptr at block row). B: [N][K] bf16 row-major (ptr at block col row).
// LDS double-buffered staging filled with async b128 copies; WMMA reads via ds_load_b128.

#define APAD 40   // 32 k-elems padded to 40 (80B row) -> conflict-free frag reads

__device__ __forceinline__ void gemm_block_128x64(const u16* __restrict__ A,
                                                  const u16* __restrict__ B,
                                                  int K,
                                                  u16* sA,   // [2][128][APAD]
                                                  u16* sB,   // [2][64][APAD]
                                                  v8f acc[4]) {
    const int t    = threadIdx.x;
    const int lane = t & 31;
    const int wv   = t >> 5;
    const int nsteps = K / 32;

    auto issue = [&](int buf, int kb) {
        u16* a  = sA + buf * 128 * APAD;
        u16* bb = sB + buf * 64 * APAD;
        // A tile: 128 rows x 64B = 512 x 16B chunks, 2 per thread
        int id = t * 2;
#pragma unroll
        for (int u = 0; u < 2; ++u, ++id) {
            int row = id >> 2, c = id & 3;
            cp16(a + row * APAD + c * 8, A + (size_t)row * K + kb + c * 8);
        }
        // B tile: 64 rows x 64B = 256 x 16B chunks, 1 per thread
        {
            int row = t >> 2, c = t & 3;
            cp16(bb + row * APAD + c * 8, B + (size_t)row * K + kb + c * 8);
        }
    };

    issue(0, 0);
    for (int s = 0; s < nsteps; ++s) {
        int cur = s & 1;
        if (s + 1 < nsteps) {
            issue(cur ^ 1, (s + 1) * 32);
            WAIT_ASYNC(3);            // 3 async ops in flight for the next buffer
        } else {
            WAIT_ASYNC(0);
        }
        __syncthreads();              // current buffer visible to all waves

        const u16* a  = sA + cur * 128 * APAD + wv * 16 * APAD;
        const u16* bb = sB + cur * 64 * APAD;
        v16bf af = load_frag(a, APAD, lane);
#pragma unroll
        for (int d = 0; d < 4; ++d) {
            v16bf bf_ = load_frag(bb + d * 16 * APAD, APAD, lane);
            acc[d] = WMMA_BF16(af, bf_, acc[d]);
        }
        __syncthreads();              // all waves done reading before buffer reuse
    }
}

// ---------------- QKV projection: [8192,1024] @ [1024,3072] ----------------

__global__ __launch_bounds__(256) void qkv_gemm(const u16* __restrict__ xb,
                                                const u16* __restrict__ wT,
                                                u16* __restrict__ q,
                                                u16* __restrict__ kk,
                                                u16* __restrict__ vT) {
    __shared__ u16 sA[2 * 128 * APAD];
    __shared__ u16 sB[2 * 64 * APAD];

    int lane = threadIdx.x & 31, wv = threadIdx.x >> 5;
    int mt = blockIdx.x / 48;         // 64 M-blocks of 128
    int nt = blockIdx.x % 48;         // 48 N-blocks of 64

    v8f acc[4] = {};
    gemm_block_128x64(xb + (size_t)mt * 128 * DIMX, wT + (size_t)nt * 64 * DIMX,
                      DIMX, sA, sB, acc);

    int half = lane >> 4, cl = lane & 15;
#pragma unroll
    for (int d = 0; d < 4; ++d) {
#pragma unroll
        for (int r = 0; r < 8; ++r) {
            int grow = mt * 128 + wv * 16 + r + 8 * half;  // [0,8192)
            int gcol = nt * 64 + d * 16 + cl;              // [0,3072)
            int b = grow >> 11, n = grow & (SEQ - 1);
            u16 bv = f2bf(acc[d][r]);
            if (gcol < DIMX) {
                int h = gcol >> 6, dh = gcol & 63;
                q[((size_t)(b * HEADS + h) * SEQ + n) * DH + dh] = bv;
            } else if (gcol < 2 * DIMX) {
                int c = gcol - DIMX;
                int h = c >> 6, dh = c & 63;
                kk[((size_t)(b * HEADS + h) * SEQ + n) * DH + dh] = bv;
            } else {
                int c = gcol - 2 * DIMX;
                int h = c >> 6, dh = c & 63;
                vT[((size_t)(b * HEADS + h) * DH + dh) * SEQ + n] = bv;
            }
        }
    }
}

// ---------------- attention: block = 4 waves on one head, K/V staged via async LDS ----------------

#define KPAD 72   // 64 dh-elems padded to 72 (144B row)

__global__ __launch_bounds__(128) void attn_kernel(const u16* __restrict__ q,
                                                   const u16* __restrict__ k,
                                                   const u16* __restrict__ vT,
                                                   u16* __restrict__ aout) {
    __shared__ u16 sK[2][32][KPAD];       // 32 keys x 64 dh (padded)
    __shared__ u16 sV[2][64][APAD];       // 64 dh rows x 32 keys (padded)
    __shared__ u16 lds_p[4][16][APAD];    // per-wave probability re-layout

    int t = threadIdx.x, lane = t & 31, wv = t >> 5;
    int bh = blockIdx.x >> 5;             // 64 head-batches
    int qg = blockIdx.x & 31;             // 32 query groups of 64
    int qt = qg * 4 + wv;                 // this wave's 16-query tile

    const u16* qb  = q  + ((size_t)bh * SEQ + qt * 16) * DH;
    const u16* kb0 = k  + (size_t)bh * SEQ * DH;
    const u16* vb0 = vT + (size_t)bh * DH * SEQ;

    v16bf aq0 = load_frag(qb,      DH, lane);   // k-dim 0..31
    v16bf aq1 = load_frag(qb + 32, DH, lane);   // k-dim 32..63

    float m[8], l[8];
    v8f o[4] = {};
#pragma unroll
    for (int r = 0; r < 8; ++r) { m[r] = -3.0e38f; l[r] = 0.0f; }

    const float sc = 0.125f;              // 64^-0.5
    int half = lane >> 4, cl = lane & 15;

    auto issue = [&](int buf, int jc) {
        // K chunk: 32 keys x 128B = 256 x 16B, 2 per thread
        int id = t * 2;
#pragma unroll
        for (int u = 0; u < 2; ++u, ++id) {
            int kr = id >> 3, c = id & 7;
            cp16(&sK[buf][kr][c * 8], kb0 + (size_t)(jc * 32 + kr) * DH + c * 8);
        }
        // V chunk: 64 dh-rows x 64B = 256 x 16B, 2 per thread
        id = t * 2;
#pragma unroll
        for (int u = 0; u < 2; ++u, ++id) {
            int dr = id >> 2, c = id & 3;
            cp16(&sV[buf][dr][c * 8], vb0 + (size_t)dr * SEQ + jc * 32 + c * 8);
        }
    };

    issue(0, 0);
    for (int jc = 0; jc < SEQ / 32; ++jc) {
        int cur = jc & 1;
        if (jc + 1 < SEQ / 32) {
            issue(cur ^ 1, jc + 1);
            WAIT_ASYNC(4);
        } else {
            WAIT_ASYNC(0);
        }
        __syncthreads();

        const u16* kc = &sK[cur][0][0];
        v8f z = {};
        v8f s0 = WMMA_BF16(aq0, load_frag(kc, KPAD, lane), z);
        s0 = WMMA_BF16(aq1, load_frag(kc + 32, KPAD, lane), s0);
        v8f s1 = WMMA_BF16(aq0, load_frag(kc + 16 * KPAD, KPAD, lane), z);
        s1 = WMMA_BF16(aq1, load_frag(kc + 16 * KPAD + 32, KPAD, lane), s1);

        float p0[8], p1[8];
#pragma unroll
        for (int r = 0; r < 8; ++r) {
            p0[r] = s0[r] * sc;
            p1[r] = s1[r] * sc;
            float mx = fmaxf(p0[r], p1[r]);
#pragma unroll
            for (int off = 1; off < 16; off <<= 1)
                mx = fmaxf(mx, __shfl_xor(mx, off, 16));
            float mn = fmaxf(m[r], mx);
            float corr = __expf(m[r] - mn);
            m[r] = mn;
            p0[r] = __expf(p0[r] - mn);
            p1[r] = __expf(p1[r] - mn);
            float rs = p0[r] + p1[r];
#pragma unroll
            for (int off = 1; off < 16; off <<= 1)
                rs += __shfl_xor(rs, off, 16);
            l[r] = l[r] * corr + rs;
#pragma unroll
            for (int d = 0; d < 4; ++d) o[d][r] *= corr;
        }

        // C-format f32 -> A-format bf16 via per-wave LDS (same-wave, in-order DS)
#pragma unroll
        for (int r = 0; r < 8; ++r) {
            lds_p[wv][r + 8 * half][cl]      = f2bf(p0[r]);
            lds_p[wv][r + 8 * half][16 + cl] = f2bf(p1[r]);
        }
        v16bf ap = load_frag(&lds_p[wv][0][0], APAD, lane);

        const u16* vc = &sV[cur][0][0];
#pragma unroll
        for (int d = 0; d < 4; ++d) {
            o[d] = WMMA_BF16(ap, load_frag(vc + d * 16 * APAD, APAD, lane), o[d]);
        }
        __syncthreads();
    }

    int b = bh >> 4, h = bh & 15;
#pragma unroll
    for (int r = 0; r < 8; ++r) {
        float inv = 1.0f / l[r];
        int row = b * SEQ + qt * 16 + r + 8 * half;
#pragma unroll
        for (int d = 0; d < 4; ++d) {
            aout[(size_t)row * DIMX + h * 64 + d * 16 + cl] = f2bf(o[d][r] * inv);
        }
    }
}

// ---------------- output projection: [8192,1024] @ [1024,1024] + bias ----------------

__global__ __launch_bounds__(256) void out_gemm(const u16* __restrict__ aout,
                                                const u16* __restrict__ wT,
                                                const float* __restrict__ bias,
                                                float* __restrict__ out) {
    __shared__ u16 sA[2 * 128 * APAD];
    __shared__ u16 sB[2 * 64 * APAD];

    int lane = threadIdx.x & 31, wv = threadIdx.x >> 5;
    int mt = blockIdx.x >> 4;         // 64
    int nt = blockIdx.x & 15;         // 16

    v8f acc[4] = {};
    gemm_block_128x64(aout + (size_t)mt * 128 * DIMX, wT + (size_t)nt * 64 * DIMX,
                      DIMX, sA, sB, acc);

    int half = lane >> 4, cl = lane & 15;
#pragma unroll
    for (int d = 0; d < 4; ++d) {
#pragma unroll
        for (int r = 0; r < 8; ++r) {
            int grow = mt * 128 + wv * 16 + r + 8 * half;
            int gcol = nt * 64 + d * 16 + cl;
            out[(size_t)grow * DIMX + gcol] = acc[d][r] + bias[gcol];
        }
    }
}

// ---------------- host launch ----------------

extern "C" void kernel_launch(void* const* d_in, const int* in_sizes, int n_in,
                              void* d_out, int out_size, void* d_ws, size_t ws_size,
                              hipStream_t stream) {
    const float* x     = (const float*)d_in[0];   // [4,2048,1024]
    const float* w_qkv = (const float*)d_in[1];   // [1024,3072]
    const float* w_out = (const float*)d_in[2];   // [1024,1024]
    const float* b_out = (const float*)d_in[3];   // [1024]
    float* out = (float*)d_out;

    u16* base  = (u16*)d_ws;
    u16* xb    = base;                                   // 8192*1024
    u16* wqkvT = xb    + (size_t)ROWS * DIMX;            // 3072*1024
    u16* woutT = wqkvT + (size_t)3 * DIMX * DIMX;        // 1024*1024
    u16* qws   = woutT + (size_t)DIMX * DIMX;            // 64*2048*64
    u16* kws   = qws   + (size_t)BHN * SEQ * DH;
    u16* vTws  = kws   + (size_t)BHN * SEQ * DH;
    u16* aows  = vTws  + (size_t)BHN * SEQ * DH;         // 8192*1024

    {
        int n4 = (ROWS * DIMX) / 4;
        cvt_bf16_vec4<<<(n4 + 255) / 256, 256, 0, stream>>>(x, xb, n4);
    }
    {
        int n = DIMX * 3 * DIMX;
        transpose_cvt<<<(n + 255) / 256, 256, 0, stream>>>(w_qkv, wqkvT, DIMX, 3 * DIMX);
        int n2 = DIMX * DIMX;
        transpose_cvt<<<(n2 + 255) / 256, 256, 0, stream>>>(w_out, woutT, DIMX, DIMX);
    }
    qkv_gemm<<<64 * 48, 256, 0, stream>>>(xb, wqkvT, qws, kws, vTws);
    attn_kernel<<<64 * 32, 128, 0, stream>>>(qws, kws, vTws, aows);
    out_gemm<<<64 * 16, 256, 0, stream>>>(aows, woutT, b_out, out);
}
